// SelfAttentionBlock_4526895530584
// MI455X (gfx1250) — matrix operand order repaired and checked by reading, not compiled, not run on hardware
//
#include <hip/hip_runtime.h>
#include <hip/hip_bf16.h>
#include <math.h>

typedef __attribute__((ext_vector_type(16))) _Float16 v16h;
typedef __attribute__((ext_vector_type(8)))  _Float16 v8h;
typedef __attribute__((ext_vector_type(8)))  float    v8f;
typedef __attribute__((ext_vector_type(4)))  unsigned int v4u;
typedef __attribute__((ext_vector_type(8)))  int v8i;
typedef __attribute__((ext_vector_type(4)))  int v4i;

#define NB   2
#define NQ   4096
#define NK   16384
#define DD   64
#define BN   64     // k-chunk width
#define NCH  (NK / BN)
#define LDK  72     // LDS row stride (halves): 128B data + 16B pad per row

// Workspace layout (bytes)
#define OFF_Q   0x000000u   // [2][4096][64] f16  = 1 MB
#define OFF_K   0x100000u   // [2][16384][64] f16 = 4 MB
#define OFF_VT  0x500000u   // [2][64][16384] f16 = 4 MB (V transposed)
#define OFF_M   0x900000u   // [2][256] f32 per-wave running max
#define OFF_Z   0x900800u   // [2][256] f32 per-wave partial Z
#define OFF_MG  0x901000u   // [2] f32 global max
#define OFF_ZG  0x901100u   // [2] f32 global Z
#define OFF_PEQ 0x901200u   // [5][64] f32 PE projection (q side)
#define OFF_PEK 0x901800u   // [5][64] f32 PE projection (k side)

typedef __attribute__((address_space(3))) const void lds_cv;
__device__ __forceinline__ unsigned lds_offset_of(const void* p) {
  return (unsigned)(unsigned long long)(lds_cv*)p;
}

// TDM: load a 64-row x 128-byte tile (8-byte elements, row stride in elements),
// inserting 16B of LDS padding after every 128B row (pad_interval=32dw, pad_amount=4dw).
__device__ __forceinline__ void tdm_load_tile(unsigned lds_off, const _Float16* gptr,
                                              int stride_elems) {
  unsigned long long ga = (unsigned long long)(uintptr_t)gptr;
  v4u g0;
  g0[0] = 1u;                                           // count=1, user mode
  g0[1] = lds_off;                                      // lds_addr (bytes)
  g0[2] = (unsigned)(ga & 0xffffffffu);                 // global_addr[31:0]
  g0[3] = (unsigned)((ga >> 32) & 0x01ffffffu) | (2u << 30);  // addr[56:32] | type=2
  v8i g1;
  g1[0] = (3 << 16) | (1 << 20) | (4 << 22) | (3 << 25); // data_size=8B, pad_en, intv=32dw, amt=4dw
  g1[1] = (int)(16u << 16);        // tensor_dim0 = 16 elems (128B row)
  g1[2] = (int)(64u << 16);        // tensor_dim0 hi=0 | tensor_dim1 = 64 rows
  g1[3] = (int)(16u << 16);        // tensor_dim1 hi=0 | tile_dim0 = 16
  g1[4] = 64;                      // tile_dim1 = 64 | tile_dim2 = 0
  g1[5] = stride_elems;            // tensor_dim0_stride[31:0] (8B units)
  g1[6] = 0;
  g1[7] = 0;
  v4i z4 = {};
  v8i z8 = {};
  __builtin_amdgcn_tensor_load_to_lds(g0, g1, z4, z4, z8, 0);
}

__device__ __forceinline__ v16h ld_frag16(const _Float16* p0, const _Float16* p1) {
  v8h lo = *(const v8h*)p0;
  v8h hi = *(const v8h*)p1;
  v16h r;
#pragma unroll
  for (int i = 0; i < 8; ++i) { r[i] = lo[i]; r[i + 8] = hi[i]; }
  return r;
}

// ---- PE projection: A=sum w[0..15], B=w[16], C=w[17], E=w[18], F=sum w[19..63]+bias
__global__ __launch_bounds__(64)
void pe_proj_kernel(const float* __restrict__ w, const float* __restrict__ bias,
                    float* __restrict__ out) {
  int d = threadIdx.x;
  float A = 0.f, F = bias[d];
  for (int c = 0; c < 16; ++c)  A += w[c * 64 + d];
  for (int c = 19; c < 64; ++c) F += w[c * 64 + d];
  out[0 * 64 + d] = A;
  out[1 * 64 + d] = w[16 * 64 + d];
  out[2 * 64 + d] = w[17 * 64 + d];
  out[3 * 64 + d] = w[18 * 64 + d];
  out[4 * 64 + d] = F;
}

// ---- Q projection: q = zx_p @ q_w + q_b + pe, scaled by 1/sqrt(D), -> f16
__global__ __launch_bounds__(64)
void proj_q_kernel(const float* __restrict__ zx, const float* __restrict__ qw,
                   const float* __restrict__ qb, const float* __restrict__ pe,
                   _Float16* __restrict__ qf) {
  int n = blockIdx.x & (NQ - 1);
  int d = threadIdx.x;
  int h = n >> 6, w = n & 63;
  int b = blockIdx.x >> 12;
  const float* zp = zx + (size_t)b * 64 * NQ + n;
  float acc = qb[d];
#pragma unroll 8
  for (int c = 0; c < 64; ++c) acc += zp[c * NQ] * qw[c * 64 + d];
  float sh = sinf(4.f * h), ch = cosf(4.f * h);
  float sw = sinf(4.f * w), cw = cosf(4.f * w);
  acc += sh * pe[d] + ch * pe[64 + d] + sw * pe[128 + d] + cw * pe[192 + d] + pe[256 + d];
  qf[(size_t)blockIdx.x * 64 + d] = (_Float16)(acc * 0.125f);
}

// ---- K/V projection: k = zy_p @ k_w + k_b + pe ; v = zy_p @ v_w + v_b (stored transposed)
__global__ __launch_bounds__(64)
void proj_kv_kernel(const float* __restrict__ zy, const float* __restrict__ kw,
                    const float* __restrict__ kb, const float* __restrict__ pek,
                    const float* __restrict__ vw, const float* __restrict__ vb,
                    _Float16* __restrict__ kf, _Float16* __restrict__ vtf) {
  int n  = blockIdx.x & 4095;
  int kk = (blockIdx.x >> 12) & 3;
  int b  = blockIdx.x >> 14;
  int d  = threadIdx.x;
  int h = n >> 6, w = n & 63;
  const float* zp = zy + ((size_t)(b * 4 + kk) * 64) * 4096 + n;
  float ka = kb[d], va = vb[d];
#pragma unroll 8
  for (int c = 0; c < 64; ++c) {
    float zv = zp[c * 4096];
    ka += zv * kw[c * 64 + d];
    va += zv * vw[c * 64 + d];
  }
  float sh = sinf(4.f * h), ch = cosf(4.f * h);
  float sw = sinf(4.f * w), cw = cosf(4.f * w);
  ka += sh * pek[d] + ch * pek[64 + d] + sw * pek[128 + d] + cw * pek[192 + d] + pek[256 + d];
  int nk = kk * 4096 + n;
  kf[((size_t)(b * NK + nk)) * 64 + d] = (_Float16)ka;
  vtf[((size_t)(b * 64 + d)) * NK + nk] = (_Float16)va;
}

// ---- Flash attention main pass: TDM double-buffered K/V staging + WMMA core
__global__ __launch_bounds__(128)
void attn_fwd_kernel(const _Float16* __restrict__ qf, const _Float16* __restrict__ kf,
                     const _Float16* __restrict__ vtf, float* __restrict__ out,
                     float* __restrict__ mbuf, float* __restrict__ zbuf) {
  __shared__ _Float16 klds[2][BN][LDK];    // K tile  [kcol][d], double buffered
  __shared__ _Float16 vlds[2][DD][LDK];    // V^T tile [d][kcol], double buffered
  __shared__ _Float16 plds[4][16][LDK];    // per-wave P staging [qrow][kcol]

  const int tid  = threadIdx.x;
  const int wave = tid >> 5;
  const int lane = tid & 31;
  const int nn   = lane & 15;
  const int hi   = lane >> 4;
  const int koff = hi * 8;
  const int b    = blockIdx.x >> 6;
  const int qblk = blockIdx.x & 63;
  const int qbase = qblk * 64 + wave * 16;

  const _Float16* kbase_g = kf  + (size_t)b * NK * DD;
  const _Float16* vbase_g = vtf + (size_t)b * DD * NK;

  // Q A-fragments (K = 0..31 and 32..63), ISA 16-bit A layout
  const _Float16* qrow = qf + ((size_t)(b * NQ + qbase + nn)) * DD;
  v16h A0 = ld_frag16(qrow + koff,      qrow + koff + 16);
  v16h A1 = ld_frag16(qrow + 32 + koff, qrow + 32 + koff + 16);

  v8f O[4] = {};
  float m = -INFINITY;
  float z = 0.f;

  // Prologue: prefetch chunks 0 and 1 via TDM (one wave issues, DMA fills LDS)
  if (wave == 0) {
    tdm_load_tile(lds_offset_of(&klds[0][0][0]), kbase_g,                16);
    tdm_load_tile(lds_offset_of(&vlds[0][0][0]), vbase_g,                4096);
    tdm_load_tile(lds_offset_of(&klds[1][0][0]), kbase_g + (size_t)BN * DD, 16);
    tdm_load_tile(lds_offset_of(&vlds[1][0][0]), vbase_g + BN,           4096);
  }

  for (int ic = 0; ic < NCH; ++ic) {
    const int buf = ic & 1;
    if (wave == 0) {
      if (ic + 1 < NCH) __builtin_amdgcn_s_wait_tensorcnt(2);  // pair(ic) done, pair(ic+1) in flight
      else              __builtin_amdgcn_s_wait_tensorcnt(0);
    }
    __syncthreads();

    // S = Q K^T : four 16-wide column subtiles
    v8f S[4];
#pragma unroll
    for (int kt = 0; kt < 4; ++kt) {
      const _Float16* br = &klds[buf][kt * 16 + nn][hi * 8];
      v16h B0 = ld_frag16(br,      br + 16);
      v16h B1 = ld_frag16(br + 32, br + 48);
      v8f acc = {};
      acc = __builtin_amdgcn_wmma_f32_16x16x32_f16(false, A0, false, B0, (short)0, acc, false, false);
      acc = __builtin_amdgcn_wmma_f32_16x16x32_f16(false, A1, false, B1, (short)0, acc, false, false);
      S[kt] = acc;
    }

    // chunk max -> wave max
    float cmax = -INFINITY;
#pragma unroll
    for (int kt = 0; kt < 4; ++kt)
#pragma unroll
      for (int i = 0; i < 8; ++i) cmax = fmaxf(cmax, S[kt][i]);
#pragma unroll
    for (int off = 16; off > 0; off >>= 1)
      cmax = fmaxf(cmax, __shfl_xor(cmax, off, 32));

    float mnew = fmaxf(m, cmax);
    float sf = __expf(m - mnew);
    z *= sf;
#pragma unroll
    for (int dt = 0; dt < 4; ++dt) O[dt] *= sf;
    m = mnew;

    // P = exp(S - m) -> f16 staging (re-layout C/D -> A via LDS)
#pragma unroll
    for (int kt = 0; kt < 4; ++kt) {
#pragma unroll
      for (int i = 0; i < 8; ++i) {
        float p = __expf(S[kt][i] - mnew);
        z += p;
        plds[wave][i + 8 * hi][kt * 16 + nn] = (_Float16)p;
      }
    }
    asm volatile("s_wait_dscnt 0" ::: "memory");

    // O += P @ V (Kdim = kcols, two 32-wide halves)
#pragma unroll
    for (int g = 0; g < 2; ++g) {
      const _Float16* pr = &plds[wave][nn][g * 32 + koff];
      v16h Ap = ld_frag16(pr, pr + 16);
#pragma unroll
      for (int dt = 0; dt < 4; ++dt) {
        const _Float16* vr = &vlds[buf][dt * 16 + nn][g * 32 + hi * 8];
        v16h Bv = ld_frag16(vr, vr + 16);
        O[dt] = __builtin_amdgcn_wmma_f32_16x16x32_f16(false, Ap, false, Bv, (short)0, O[dt], false, false);
      }
    }
    __syncthreads();

    // Prefetch chunk ic+2 into the buffer we just finished reading
    if (wave == 0 && ic + 2 < NCH) {
      tdm_load_tile(lds_offset_of(&klds[buf][0][0]),
                    kbase_g + (size_t)(ic + 2) * BN * DD, 16);
      tdm_load_tile(lds_offset_of(&vlds[buf][0][0]),
                    vbase_g + (ic + 2) * BN, 4096);
    }
  }

  // write unnormalized O (C/D f32 layout: M = i + 8*hi, N = nn)
#pragma unroll
  for (int dt = 0; dt < 4; ++dt)
#pragma unroll
    for (int i = 0; i < 8; ++i) {
      int q = qbase + i + 8 * hi;
      out[((size_t)(b * NQ + q)) * DD + dt * 16 + nn] = O[dt][i];
    }

  float zt = z;
#pragma unroll
  for (int off = 16; off > 0; off >>= 1) zt += __shfl_xor(zt, off, 32);
  if (lane == 0) {
    int idx = b * 256 + qblk * 4 + wave;
    mbuf[idx] = m;
    zbuf[idx] = zt;
  }
}

// ---- merge per-wave (m, z) into per-batch (M, Z)
__global__ __launch_bounds__(256)
void reduce_mz_kernel(const float* __restrict__ mbuf, const float* __restrict__ zbuf,
                      float* __restrict__ mg, float* __restrict__ zg) {
  __shared__ float sm[256];
  __shared__ float sz[256];
  int t = threadIdx.x, b = blockIdx.x;
  float mi = mbuf[b * 256 + t];
  sm[t] = mi;
  __syncthreads();
  for (int s = 128; s > 0; s >>= 1) {
    if (t < s) sm[t] = fmaxf(sm[t], sm[t + s]);
    __syncthreads();
  }
  float M = sm[0];
  __syncthreads();
  sz[t] = zbuf[b * 256 + t] * __expf(mi - M);
  __syncthreads();
  for (int s = 128; s > 0; s >>= 1) {
    if (t < s) sz[t] += sz[t + s];
    __syncthreads();
  }
  if (t == 0) { mg[b] = M; zg[b] = sz[0]; }
}

// ---- final rescale: out *= exp(m_wave - M) / Z
__global__ __launch_bounds__(256)
void scale_out_kernel(float* __restrict__ out, const float* __restrict__ mbuf,
                      const float* __restrict__ mg, const float* __restrict__ zg) {
  int idx = blockIdx.x * 256 + threadIdx.x;
  int b = idx >> 18;             // 4096*64 = 2^18
  int q = (idx >> 6) & 4095;
  float f = __expf(mbuf[b * 256 + (q >> 4)] - mg[b]) / zg[b];
  out[idx] = out[idx] * f;
}

extern "C" void kernel_launch(void* const* d_in, const int* in_sizes, int n_in,
                              void* d_out, int out_size, void* d_ws, size_t ws_size,
                              hipStream_t stream) {
  (void)in_sizes; (void)n_in; (void)out_size; (void)ws_size;
  const float* zx   = (const float*)d_in[0];
  const float* zy   = (const float*)d_in[1];
  const float* q_w  = (const float*)d_in[2];
  const float* q_b  = (const float*)d_in[3];
  const float* px_w = (const float*)d_in[4];
  const float* px_b = (const float*)d_in[5];
  const float* k_w  = (const float*)d_in[6];
  const float* k_b  = (const float*)d_in[7];
  const float* py_w = (const float*)d_in[8];
  const float* py_b = (const float*)d_in[9];
  const float* v_w  = (const float*)d_in[10];
  const float* v_b  = (const float*)d_in[11];
  float* out = (float*)d_out;

  char* ws = (char*)d_ws;
  _Float16* qf   = (_Float16*)(ws + OFF_Q);
  _Float16* kf   = (_Float16*)(ws + OFF_K);
  _Float16* vtf  = (_Float16*)(ws + OFF_VT);
  float*    mbuf = (float*)(ws + OFF_M);
  float*    zbuf = (float*)(ws + OFF_Z);
  float*    mg   = (float*)(ws + OFF_MG);
  float*    zg   = (float*)(ws + OFF_ZG);
  float*    peq  = (float*)(ws + OFF_PEQ);
  float*    pek  = (float*)(ws + OFF_PEK);

  pe_proj_kernel<<<1, 64, 0, stream>>>(px_w, px_b, peq);
  pe_proj_kernel<<<1, 64, 0, stream>>>(py_w, py_b, pek);
  proj_q_kernel<<<NB * NQ, 64, 0, stream>>>(zx, q_w, q_b, peq, qf);
  proj_kv_kernel<<<NB * 4 * 4096, 64, 0, stream>>>(zy, k_w, k_b, pek, v_w, v_b, kf, vtf);
  attn_fwd_kernel<<<NB * 64, 128, 0, stream>>>(qf, kf, vtf, out, mbuf, zbuf);
  reduce_mz_kernel<<<NB, 256, 0, stream>>>(mbuf, zbuf, mg, zg);
  scale_out_kernel<<<(NB * NQ * DD) / 256, 256, 0, stream>>>(out, mbuf, mg, zg);
}
